// Mamba2_42649025249466
// MI455X (gfx1250) — compile-verified, matrix operand order checked by reading, not run on hardware
//
#include <hip/hip_runtime.h>
#include <hip/hip_bf16.h>

// ---------------------------------------------------------------------------
// Mamba2 forward for MI455X (gfx1250).  All GEMM-shaped work runs on
// v_wmma_f32_16x16x32_bf16 (fp32 accumulate); the two projection GEMMs stage
// their tiles with the CDNA5 Tensor Data Mover (tensor_load_to_lds, TENSORcnt)
// in a double-buffered DMA/WMMA pipeline.  wave32 layouts per ISA 7.12.2.
// ---------------------------------------------------------------------------

typedef __attribute__((ext_vector_type(16))) __bf16 v16bf;
typedef __attribute__((ext_vector_type(8)))  float  v8f;
typedef __attribute__((ext_vector_type(4)))  unsigned u32x4;
typedef __attribute__((ext_vector_type(8)))  unsigned u32x8;

#define DEV __device__ __forceinline__

constexpr int B_     = 2;
constexpr int L_     = 2048;
constexpr int DMODEL = 2048;
constexpr int INTERD = 4096;
constexpr int NHEAD  = 64;
constexpr int HDIM   = 64;
constexpr int NSTATE = 128;
constexpr int CSZ    = 256;
constexpr int NCHUNK = 8;
constexpr int CONVD  = 4352;   // INTER + 2*G*N
constexpr int PROJD  = 8512;   // INTER + CONVD + NH
constexpr int BLROWS = B_ * L_;  // 4096

// ----------------------------- bf16 helpers --------------------------------
DEV unsigned short f2bfbits(float f) {
  unsigned u = __builtin_bit_cast(unsigned, f);
  return (unsigned short)((u + 0x7FFFu + ((u >> 16) & 1u)) >> 16);
}
DEV __bf16 f2bf(float f) { return __builtin_bit_cast(__bf16, f2bfbits(f)); }
DEV float bfbits2f(unsigned short s) {
  return __builtin_bit_cast(float, ((unsigned)s) << 16);
}
DEV float bf2f(__bf16 b) { return bfbits2f(__builtin_bit_cast(unsigned short, b)); }

DEV v8f zero8() { v8f v = {0.f,0.f,0.f,0.f,0.f,0.f,0.f,0.f}; return v; }

DEV v8f wmma_bf16(v16bf a, v16bf b, v8f c) {
  return __builtin_amdgcn_wmma_f32_16x16x32_bf16(false, a, false, b,
                                                 (short)0, c, false, false);
}

// Fragment load from LDS, row-major bf16 [row][k], 16x32 operand.
// Lane layout (ISA 7.12.2, 16-bit A 16x32): lanes 0-15 row=lane hold
// K {kh..kh+7, kh+16..kh+23} with kh = (lane>>4)*8.
DEV v16bf frag_ld(const __bf16* lds, int row0, int stride, int k0, int lane) {
  const __bf16* p = lds + (row0 + (lane & 15)) * stride + k0 + ((lane >> 4) << 3);
  v16bf v;
#pragma unroll
  for (int i = 0; i < 8; ++i) { v[i] = p[i]; v[i + 8] = p[16 + i]; }
  return v;
}

// Fragment load from fp32 LDS tile (TDM-staged), converting to bf16 in-flight.
DEV v16bf frag_ld_f32(const float* lds, int row0, int stride, int k0, int lane) {
  const float* p = lds + (row0 + (lane & 15)) * stride + k0 + ((lane >> 4) << 3);
  v16bf v;
#pragma unroll
  for (int i = 0; i < 8; ++i) { v[i] = f2bf(p[i]); v[i + 8] = f2bf(p[16 + i]); }
  return v;
}

// Same as frag_ld, each element multiplied by a per-K fp32 weight (decay*dt).
DEV v16bf frag_ld_scaled(const __bf16* lds, int row0, int stride, int k0,
                         const float* w, int lane) {
  int kh = (lane >> 4) << 3;
  const __bf16* p = lds + (row0 + (lane & 15)) * stride + k0 + kh;
  const float* wp = w + k0 + kh;
  v16bf v;
#pragma unroll
  for (int i = 0; i < 8; ++i) {
    v[i]     = f2bf(bf2f(p[i]) * wp[i]);
    v[i + 8] = f2bf(bf2f(p[16 + i]) * wp[16 + i]);
  }
  return v;
}

// Masked-decay A-fragment: T[i][j] = S[i][j]*exp(acum[i]-acum[j])*dt[j] (j<=i).
DEV v16bf frag_ld_T(const unsigned short* S, const float* acum, const float* dt,
                    int row0, int j0, int lane) {
  int i  = row0 + (lane & 15);
  int kh = (lane >> 4) << 3;
  float ai = acum[i];
  const unsigned short* sp = S + i * CSZ + j0 + kh;
  v16bf v;
#pragma unroll
  for (int e = 0; e < 8; ++e) {
    int j1 = j0 + kh + e;
    int j2 = j0 + kh + 16 + e;
    float t1 = (j1 <= i) ? bfbits2f(sp[e])      * __expf(ai - acum[j1]) * dt[j1] : 0.f;
    float t2 = (j2 <= i) ? bfbits2f(sp[16 + e]) * __expf(ai - acum[j2]) * dt[j2] : 0.f;
    v[e]     = f2bf(t1);
    v[e + 8] = f2bf(t2);
  }
  return v;
}

// ------------------------- Tensor Data Mover -------------------------------
// Issue a 2-D TDM load of a rows x cols fp32 tile (row stride = stride elems)
// from global memory into LDS at byte offset lds_off.  Descriptor per CDNA5
// ISA 8.3/8.4: group0 = {count/flags, lds_addr, global_addr, type=2},
// group1 = {data_size, tensor dims, tile dims, dim0 stride}.
DEV void tdm_load_tile_f32(unsigned lds_off, const float* gsrc,
                           unsigned rows, unsigned cols, unsigned stride) {
  unsigned long long ga = (unsigned long long)(const void*)gsrc;
  u32x4 g0;
  g0[0] = 1u;                                            // count=1 (valid D#)
  g0[1] = lds_off;                                       // lds_addr (bytes)
  g0[2] = (unsigned)(ga & 0xFFFFFFFFull);                // global_addr[31:0]
  g0[3] = (unsigned)((ga >> 32) & 0x01FFFFFFull)         // global_addr[56:32]
        | (2u << 30);                                    // type = 2 ("image")
  u32x8 g1;
  g1[0] = 2u << 16;                                      // data_size=4B, no mcast
  g1[1] = (cols & 0xFFFFu) << 16;                        // tensor_dim0[15:0]
  g1[2] = (cols >> 16) | ((rows & 0xFFFFu) << 16);       // dim0 hi | dim1 lo
  g1[3] = (rows >> 16) | (cols << 16);                   // dim1 hi | tile_dim0
  g1[4] = rows & 0xFFFFu;                                // tile_dim1 (dim2=0)
  g1[5] = stride;                                        // dim0_stride[31:0]
  g1[6] = 0u;                                            // stride hi, dim1_stride
  g1[7] = 0u;
  asm volatile("tensor_load_to_lds %0, %1" :: "s"(g0), "s"(g1) : "memory");
}

// ------------------------- generic WMMA GEMM -------------------------------
// C[M x N] = A[M x K] @ W[N x K]^T, fp32 in memory, bf16 on the MACs.
// 256 threads = 8 waves, 64x64 output tile, K-slab 32.  Tiles are staged by
// the TDM (one 16x32 sub-tile per wave: waves 0-3 stage A, 4-7 stage W),
// double-buffered so slab s+1's DMA overlaps slab s's WMMAs.
__global__ __launch_bounds__(256) void gemm_wmma_kernel(
    const float* __restrict__ A, const float* __restrict__ W,
    float* __restrict__ C, int M, int N, int K) {
  __shared__ float sAf[2][64 * 32];
  __shared__ float sWf[2][64 * 32];
  int m0 = blockIdx.y * 64;
  int n0 = blockIdx.x * 64;
  int tid = threadIdx.x;
  int wave = tid >> 5, lane = tid & 31;
  int wm = wave & 3, wn = wave >> 2;        // 4x2 wave grid -> 16x32 per wave
  v8f acc0 = zero8(), acc1 = zero8();

  // This wave's staging role: 16 rows of A (waves 0-3) or of W (waves 4-7).
  const float* srcbase = (wave < 4)
      ? (A + (size_t)(m0 + wave * 16) * K)
      : (W + (size_t)(n0 + (wave - 4) * 16) * K);
  float* dst[2];
  dst[0] = (wave < 4) ? &sAf[0][wave * 16 * 32] : &sWf[0][(wave - 4) * 16 * 32];
  dst[1] = (wave < 4) ? &sAf[1][wave * 16 * 32] : &sWf[1][(wave - 4) * 16 * 32];

  int nslab = K >> 5;
  tdm_load_tile_f32((unsigned)(size_t)(void*)dst[0], srcbase, 16, 32,
                    (unsigned)K);
  for (int s = 0; s < nslab; ++s) {
    int buf = s & 1;
    if (s + 1 < nslab) {
      tdm_load_tile_f32((unsigned)(size_t)(void*)dst[buf ^ 1],
                        srcbase + (size_t)(s + 1) * 32, 16, 32, (unsigned)K);
      __builtin_amdgcn_s_wait_tensorcnt(1);   // slab s landed, s+1 in flight
    } else {
      __builtin_amdgcn_s_wait_tensorcnt(0);
    }
    __syncthreads();                          // slab s visible to all waves
    v16bf a  = frag_ld_f32(sAf[buf], wm * 16, 32, 0, lane);
    v16bf b0 = frag_ld_f32(sWf[buf], wn * 32, 32, 0, lane);
    v16bf b1 = frag_ld_f32(sWf[buf], wn * 32 + 16, 32, 0, lane);
    acc0 = wmma_bf16(a, b0, acc0);
    acc1 = wmma_bf16(a, b1, acc1);
    __syncthreads();                          // reads done before buf re-DMA
  }
  int col0 = n0 + wn * 32 + (lane & 15);
  int r0   = m0 + wm * 16 + ((lane >> 4) << 3);
#pragma unroll
  for (int r = 0; r < 8; ++r) {
    C[(size_t)(r0 + r) * N + col0]      = acc0[r];
    C[(size_t)(r0 + r) * N + col0 + 16] = acc1[r];
  }
}

// ---------------------- depthwise causal conv + SiLU -----------------------
__global__ __launch_bounds__(256) void conv_silu_kernel(
    const float* __restrict__ proj, const float* __restrict__ conv_w,
    const float* __restrict__ conv_b, float* __restrict__ hbc) {
  int idx = blockIdx.x * 256 + threadIdx.x;           // over B*L*CONVD
  if (idx >= BLROWS * CONVD) return;
  int ch = idx % CONVD;
  int bl = idx / CONVD;
  int l  = bl % L_;
  int bb = bl / L_;
  float acc = conv_b[ch];
#pragma unroll
  for (int t = 0; t < 4; ++t) {
    int ls = l - 3 + t;
    if (ls >= 0)
      acc += conv_w[ch * 4 + t] * proj[(size_t)(bb * L_ + ls) * PROJD + INTERD + ch];
  }
  hbc[idx] = acc / (1.f + __expf(-acc));              // silu
}

// ------------------------------ dt softplus --------------------------------
__global__ __launch_bounds__(256) void dt_kernel(
    const float* __restrict__ proj, const float* __restrict__ dt_bias,
    float* __restrict__ dtv) {
  int idx = blockIdx.x * 256 + threadIdx.x;           // over B*L*NH
  int h = idx & (NHEAD - 1);
  int bl = idx >> 6;
  float v = proj[(size_t)bl * PROJD + INTERD + CONVD + h] + dt_bias[h];
  dtv[idx] = (v > 20.f) ? v : logf(1.f + __expf(v));
}

// --------------------------- S = C @ B^T per (b,c) -------------------------
// 128x64 block tile per launch block; upper-triangular-only blocks skipped.
__global__ __launch_bounds__(256) void s_kernel(
    const float* __restrict__ hbc, unsigned short* __restrict__ S) {
  int cb = blockIdx.x;          // col block (64 j)
  int rb = blockIdx.y;          // row block (128 i)
  int bc = blockIdx.z;          // b*NCHUNK + c
  if (cb * 64 > rb * 128 + 127) return;   // fully masked region
  __shared__ __bf16 sC[128 * 136];
  __shared__ __bf16 sB[64 * 136];
  int tid = threadIdx.x;
  size_t rowbase = (size_t)((bc >> 3) * L_ + (bc & 7) * CSZ);
  {
    int i = tid >> 1, k0 = (tid & 1) * 64;
    const float* src = hbc + (rowbase + rb * 128 + i) * CONVD + (INTERD + NSTATE) + k0;
    __bf16* dst = sC + i * 136 + k0;
#pragma unroll 8
    for (int k = 0; k < 64; ++k) dst[k] = f2bf(src[k]);
  }
  {
    int j = tid >> 2, k0 = (tid & 3) * 32;
    const float* src = hbc + (rowbase + cb * 64 + j) * CONVD + INTERD + k0;
    __bf16* dst = sB + j * 136 + k0;
#pragma unroll 8
    for (int k = 0; k < 32; ++k) dst[k] = f2bf(src[k]);
  }
  __syncthreads();
  int wave = tid >> 5, lane = tid & 31;   // wave = 16-row stripe
  v8f acc[4] = {zero8(), zero8(), zero8(), zero8()};
  for (int kt = 0; kt < 128; kt += 32) {
    v16bf a = frag_ld(sC, wave * 16, 136, kt, lane);
#pragma unroll
    for (int t = 0; t < 4; ++t) {
      v16bf b = frag_ld(sB, t * 16, 136, kt, lane);
      acc[t] = wmma_bf16(a, b, acc[t]);
    }
  }
  unsigned short* out = S + (size_t)bc * (CSZ * CSZ);
  int row0 = rb * 128 + wave * 16 + ((lane >> 4) << 3);
  int colb = cb * 64 + (lane & 15);
#pragma unroll
  for (int t = 0; t < 4; ++t)
#pragma unroll
    for (int r = 0; r < 8; ++r)
      out[(size_t)(row0 + r) * CSZ + colb + t * 16] = f2bfbits(acc[t][r]);
}

// --------------------- intra-chunk kernel per (b,c,h) ----------------------
// Phase 1: cumsum(dA); Phase 2: Ydiag = (S .* L .* dt) @ x  (+ x*D);
// Phase 3: chunk states = (x*w)^T @ B.
__global__ __launch_bounds__(256) void chunk_kernel(
    const float* __restrict__ hbc, const float* __restrict__ dtv,
    const unsigned short* __restrict__ S, const float* __restrict__ A_log,
    const float* __restrict__ Dv, float* __restrict__ Y,
    float* __restrict__ statesW, float* __restrict__ acumW) {
  int c = blockIdx.x, h = blockIdx.y, b = blockIdx.z;
  __shared__ __bf16 xT[64 * 264];     // x transposed [p][j], bf16
  __shared__ __bf16 BTs[128 * 40];    // B^T K-slab  [n][j-jt*32]
  __shared__ float dtl[256], acum[256], w[256];
  int tid = threadIdx.x;
  size_t rowbase = (size_t)(b * L_ + c * CSZ);

  float Ah  = -__expf(A_log[h]);
  float dti = dtv[(rowbase + tid) * NHEAD + h];
  dtl[tid]  = dti;
  acum[tid] = dti * Ah;
  __syncthreads();
  for (int off = 1; off < 256; off <<= 1) {     // Hillis-Steele inclusive scan
    float v = acum[tid];
    float u = (tid >= off) ? acum[tid - off] : 0.f;
    __syncthreads();
    acum[tid] = v + u;
    __syncthreads();
  }
  float alast = acum[255];
  w[tid] = __expf(alast - acum[tid]) * dtl[tid];
  acumW[(((size_t)(b * NHEAD + h)) * NCHUNK + c) * CSZ + tid] = acum[tid];
  {   // stage x transposed (coalesced global read along p)
    const float* xsrc = hbc + (rowbase + tid) * CONVD + h * HDIM;
#pragma unroll 4
    for (int p = 0; p < 64; ++p) xT[p * 264 + tid] = f2bf(xsrc[p]);
  }
  __syncthreads();

  int wave = tid >> 5, lane = tid & 31;
  const unsigned short* Sp = S + (size_t)(b * NCHUNK + c) * (CSZ * CSZ);
  float Dh = Dv[h];

  // ---- Phase 2: Ydiag, wave owns rows [wave*32, wave*32+31], cols 0..63 ----
  v8f acc[2][4];
#pragma unroll
  for (int i = 0; i < 2; ++i)
#pragma unroll
    for (int j = 0; j < 4; ++j) acc[i][j] = zero8();
  for (int jt = 0; jt <= wave; ++jt) {          // triangular: skip j > i
    v16bf a0 = frag_ld_T(Sp, acum, dtl, wave * 32,      jt * 32, lane);
    v16bf a1 = frag_ld_T(Sp, acum, dtl, wave * 32 + 16, jt * 32, lane);
#pragma unroll
    for (int ct = 0; ct < 4; ++ct) {
      v16bf bb = frag_ld(xT, ct * 16, 264, jt * 32, lane);
      acc[0][ct] = wmma_bf16(a0, bb, acc[0][ct]);
      acc[1][ct] = wmma_bf16(a1, bb, acc[1][ct]);
    }
  }
#pragma unroll
  for (int rt = 0; rt < 2; ++rt)
#pragma unroll
    for (int ct = 0; ct < 4; ++ct) {
      int col  = ct * 16 + (lane & 15);
      int row0 = wave * 32 + rt * 16 + ((lane >> 4) << 3);
#pragma unroll
      for (int r = 0; r < 8; ++r) {
        int i = row0 + r;
        float xv = bf2f(xT[col * 264 + i]);
        Y[(rowbase + i) * INTERD + h * HDIM + col] = acc[rt][ct][r] + xv * Dh;
      }
    }

  // ---- Phase 3: states[p][n] = sum_j x[j][p]*w[j]*B[j][n] -----------------
  int wm = wave & 3, wn = wave >> 2;
  v8f sacc[4] = {zero8(), zero8(), zero8(), zero8()};
  for (int jt = 0; jt < 8; ++jt) {
    __syncthreads();
    {   // stage B^T slab: rows j = jt*32+(tid>>3), 16 n per thread
      int jj = tid >> 3;
      int n0 = (tid & 7) * 16;
      const float* bsrc = hbc + (rowbase + jt * 32 + jj) * CONVD + INTERD + n0;
#pragma unroll
      for (int n = 0; n < 16; ++n) BTs[(n0 + n) * 40 + jj] = f2bf(bsrc[n]);
    }
    __syncthreads();
    v16bf a = frag_ld_scaled(xT, wm * 16, 264, jt * 32, w, lane);
#pragma unroll
    for (int ct = 0; ct < 4; ++ct) {
      v16bf bb = frag_ld(BTs, wn * 64 + ct * 16, 40, 0, lane);
      sacc[ct] = wmma_bf16(a, bb, sacc[ct]);
    }
  }
  float* stp = statesW + (size_t)((b * NCHUNK + c) * NHEAD + h) * (HDIM * NSTATE);
  int prow = wm * 16 + ((lane >> 4) << 3);
#pragma unroll
  for (int ct = 0; ct < 4; ++ct) {
    int col = wn * 64 + ct * 16 + (lane & 15);
#pragma unroll
    for (int r = 0; r < 8; ++r) stp[(prow + r) * NSTATE + col] = sacc[ct][r];
  }
}

// -------------------- inter-chunk sequential scan (tiny) -------------------
__global__ __launch_bounds__(256) void scan_kernel(
    const float* __restrict__ statesW, const float* __restrict__ acumW,
    float* __restrict__ prevW) {
  int bh = blockIdx.x;
  int b = bh >> 6, h = bh & 63;
  int tid = threadIdx.x;
  float dec[NCHUNK];
#pragma unroll
  for (int c = 0; c < NCHUNK; ++c)
    dec[c] = __expf(acumW[(((size_t)(b * NHEAD + h)) * NCHUNK + c) * CSZ + 255]);
  for (int t = 0; t < 32; ++t) {
    int idx = t * 256 + tid;                  // over 8192 (p,n) elements
    float carry = 0.f;
#pragma unroll
    for (int c = 0; c < NCHUNK; ++c) {
      size_t off = (size_t)((b * NCHUNK + c) * NHEAD + h) * (HDIM * NSTATE) + idx;
      prevW[off] = carry;
      carry = dec[c] * carry + statesW[off];
    }
  }
}

// ------------------- Yoff = exp(acum_i) * (C @ prev^T) ---------------------
__global__ __launch_bounds__(256) void yoff_kernel(
    const float* __restrict__ hbc, const float* __restrict__ prevW,
    const float* __restrict__ acumW, float* __restrict__ Y) {
  int c = blockIdx.x, h = blockIdx.y, b = blockIdx.z;
  __shared__ __bf16 prevL[64 * 136];   // [p][n]
  __shared__ __bf16 sC[256 * 40];      // C K-slab [i][n-nt*32]
  __shared__ float eA[256];
  int tid = threadIdx.x;
  size_t rowbase = (size_t)(b * L_ + c * CSZ);
  eA[tid] = __expf(acumW[(((size_t)(b * NHEAD + h)) * NCHUNK + c) * CSZ + tid]);
  {
    int p = tid >> 2, n0 = (tid & 3) * 32;
    const float* src = prevW + (size_t)((b * NCHUNK + c) * NHEAD + h) * (HDIM * NSTATE)
                     + p * NSTATE + n0;
#pragma unroll 8
    for (int n = 0; n < 32; ++n) prevL[p * 136 + n0 + n] = f2bf(src[n]);
  }
  int wave = tid >> 5, lane = tid & 31;
  v8f acc[2][4];
#pragma unroll
  for (int i = 0; i < 2; ++i)
#pragma unroll
    for (int j = 0; j < 4; ++j) acc[i][j] = zero8();
  for (int nt = 0; nt < 4; ++nt) {
    __syncthreads();
    {
      const float* src = hbc + (rowbase + tid) * CONVD + (INTERD + NSTATE) + nt * 32;
#pragma unroll 8
      for (int n = 0; n < 32; ++n) sC[tid * 40 + n] = f2bf(src[n]);
    }
    __syncthreads();
    v16bf a0 = frag_ld(sC, wave * 32,      40, 0, lane);
    v16bf a1 = frag_ld(sC, wave * 32 + 16, 40, 0, lane);
#pragma unroll
    for (int ct = 0; ct < 4; ++ct) {
      v16bf bb = frag_ld(prevL, ct * 16, 136, nt * 32, lane);
      acc[0][ct] = wmma_bf16(a0, bb, acc[0][ct]);
      acc[1][ct] = wmma_bf16(a1, bb, acc[1][ct]);
    }
  }
#pragma unroll
  for (int rt = 0; rt < 2; ++rt)
#pragma unroll
    for (int ct = 0; ct < 4; ++ct) {
      int col  = ct * 16 + (lane & 15);
      int row0 = wave * 32 + rt * 16 + ((lane >> 4) << 3);
#pragma unroll
      for (int r = 0; r < 8; ++r) {
        int i = row0 + r;
        size_t o = (rowbase + i) * INTERD + h * HDIM + col;
        Y[o] += eA[i] * acc[rt][ct][r];
      }
    }
}

// ----------------------- gate * SiLU + RMSNorm -----------------------------
__global__ __launch_bounds__(256) void gaterms_kernel(
    const float* __restrict__ Y, const float* __restrict__ proj,
    const float* __restrict__ norm_w, float* __restrict__ yn) {
  int bl = blockIdx.x;
  int tid = threadIdx.x;
  __shared__ float red[256];
  float ss = 0.f;
  for (int i = tid; i < INTERD; i += 256) {
    float g = proj[(size_t)bl * PROJD + i];
    float y = Y[(size_t)bl * INTERD + i] * (g / (1.f + __expf(-g)));
    yn[(size_t)bl * INTERD + i] = y;
    ss += y * y;
  }
  red[tid] = ss;
  __syncthreads();
  for (int s = 128; s > 0; s >>= 1) {
    if (tid < s) red[tid] += red[tid + s];
    __syncthreads();
  }
  float r = rsqrtf(red[0] / (float)INTERD + 1e-5f);
  for (int i = tid; i < INTERD; i += 256)
    yn[(size_t)bl * INTERD + i] = yn[(size_t)bl * INTERD + i] * r * norm_w[i];
}

// ----------------------------- workspace map -------------------------------
constexpr size_t alg(size_t x) { return (x + 255) & ~(size_t)255; }
constexpr size_t OFF_PROJ = 0;
constexpr size_t OFF_HBC  = OFF_PROJ + alg((size_t)BLROWS * PROJD * 4);
constexpr size_t OFF_DT   = OFF_HBC  + alg((size_t)BLROWS * CONVD * 4);
constexpr size_t OFF_S    = OFF_DT   + alg((size_t)BLROWS * NHEAD * 4);
constexpr size_t OFF_ACUM = OFF_S    + alg((size_t)B_ * NCHUNK * CSZ * CSZ * 2);
constexpr size_t OFF_ST   = OFF_ACUM + alg((size_t)B_ * NHEAD * NCHUNK * CSZ * 4);
constexpr size_t OFF_PREV = OFF_ST   + alg((size_t)B_ * NCHUNK * NHEAD * HDIM * NSTATE * 4);
constexpr size_t OFF_Y    = OFF_PREV + alg((size_t)B_ * NCHUNK * NHEAD * HDIM * NSTATE * 4);
constexpr size_t OFF_YN   = OFF_Y    + alg((size_t)BLROWS * INTERD * 4);
constexpr size_t WS_NEED  = OFF_YN   + alg((size_t)BLROWS * INTERD * 4);

extern "C" void kernel_launch(void* const* d_in, const int* in_sizes, int n_in,
                              void* d_out, int out_size, void* d_ws, size_t ws_size,
                              hipStream_t stream) {
  (void)in_sizes; (void)n_in; (void)out_size;
  if (ws_size < WS_NEED) return;   // ~397 MiB of scratch required
  const float* hidden  = (const float*)d_in[0];
  const float* W_in    = (const float*)d_in[1];
  const float* conv_w  = (const float*)d_in[2];
  const float* conv_b  = (const float*)d_in[3];
  const float* dt_bias = (const float*)d_in[4];
  const float* A_log   = (const float*)d_in[5];
  const float* Dv      = (const float*)d_in[6];
  const float* norm_w  = (const float*)d_in[7];
  const float* W_out   = (const float*)d_in[8];
  float* out = (float*)d_out;
  char* ws = (char*)d_ws;
  float*          proj    = (float*)(ws + OFF_PROJ);
  float*          hbc     = (float*)(ws + OFF_HBC);
  float*          dtv     = (float*)(ws + OFF_DT);
  unsigned short* Sw      = (unsigned short*)(ws + OFF_S);
  float*          acumW   = (float*)(ws + OFF_ACUM);
  float*          statesW = (float*)(ws + OFF_ST);
  float*          prevW   = (float*)(ws + OFF_PREV);
  float*          Yw      = (float*)(ws + OFF_Y);
  float*          ynW     = (float*)(ws + OFF_YN);

  // 1) in-projection: [4096,2048] @ [8512,2048]^T  (TDM-staged WMMA GEMM)
  gemm_wmma_kernel<<<dim3(PROJD / 64, BLROWS / 64), 256, 0, stream>>>(
      hidden, W_in, proj, BLROWS, PROJD, DMODEL);
  // 2) depthwise causal conv + SiLU over conv channels
  conv_silu_kernel<<<(BLROWS * CONVD) / 256, 256, 0, stream>>>(
      proj, conv_w, conv_b, hbc);
  // 3) dt = softplus(dt + bias)
  dt_kernel<<<(BLROWS * NHEAD) / 256, 256, 0, stream>>>(proj, dt_bias, dtv);
  // 4) S = C @ B^T per (b,c) (head-shared since G=1)
  s_kernel<<<dim3(4, 2, B_ * NCHUNK), 256, 0, stream>>>(hbc, Sw);
  // 5) intra-chunk: cumsum, Ydiag (+x*D), chunk states
  chunk_kernel<<<dim3(NCHUNK, NHEAD, B_), 256, 0, stream>>>(
      hbc, dtv, Sw, A_log, Dv, Yw, statesW, acumW);
  // 6) inter-chunk sequential scan
  scan_kernel<<<B_ * NHEAD, 256, 0, stream>>>(statesW, acumW, prevW);
  // 7) Yoff accumulation
  yoff_kernel<<<dim3(NCHUNK, NHEAD, B_), 256, 0, stream>>>(
      hbc, prevW, acumW, Yw);
  // 8) gate * SiLU + RMSNorm
  gaterms_kernel<<<BLROWS, 256, 0, stream>>>(Yw, proj, norm_w, ynW);
  // 9) out-projection: [4096,4096] @ [2048,4096]^T  (TDM-staged WMMA GEMM)
  gemm_wmma_kernel<<<dim3(DMODEL / 64, BLROWS / 64), 256, 0, stream>>>(
      ynW, W_out, out, BLROWS, DMODEL, INTERD);
}
